// DiffGRUCell_19774029431558
// MI455X (gfx1250) — matrix operand order repaired and checked by reading, not compiled, not run on hardware
//
#include <hip/hip_runtime.h>

// ---- problem constants (from reference) ----
#define BB 64
#define NN 325
#define MM (BB * NN)       // 20800 rows
#define KF 192             // K*F
#define HH 1024
#define CONCAT 1216        // KF + HH
#define KC 32              // WMMA K per step (bf16)
#define NKC (CONCAT / KC)  // 38 chunks
#define XCHUNKS (KF / KC)  // 6 chunks come from x_flat

typedef __attribute__((ext_vector_type(16))) __bf16 v16bf;
typedef __attribute__((ext_vector_type(8)))  float  v8f;
typedef __attribute__((ext_vector_type(4)))  int    v4i;

union Frag { v16bf v; uint4 q[2]; };

// ---- CDNA5 async global->LDS path (double buffered), with sync fallback ----
#if defined(__gfx1250__) && __has_builtin(__builtin_amdgcn_global_load_async_to_lds_b128)
#define USE_ASYNC 1
#else
#define USE_ASYNC 0
#endif

#if USE_ASYNC
typedef __attribute__((address_space(1))) v4i glob_v4i;
typedef __attribute__((address_space(3))) v4i lds_v4i;
#endif

__device__ __forceinline__ void copy_b128(unsigned short* lds_dst,
                                          const unsigned short* gsrc) {
#if USE_ASYNC
    __builtin_amdgcn_global_load_async_to_lds_b128(
        (glob_v4i*)gsrc, (lds_v4i*)lds_dst, /*imm offset*/ 0, /*cpol*/ 0);
#else
    *(uint4*)lds_dst = *(const uint4*)gsrc;
#endif
}

#if USE_ASYNC
#if __has_builtin(__builtin_amdgcn_s_wait_asynccnt)
#define WAIT_ASYNC(n) __builtin_amdgcn_s_wait_asynccnt(n)
#else
#define WAIT_ASYNC(n) asm volatile("s_wait_asynccnt %0" ::"n"(n) : "memory")
#endif
#else
#define WAIT_ASYNC(n)
#endif

__device__ __forceinline__ unsigned short f2bf(float f) {
    unsigned int u = __float_as_uint(f);
    u += 0x7FFFu + ((u >> 16) & 1u);   // round-to-nearest-even
    return (unsigned short)(u >> 16);
}
__device__ __forceinline__ float bf2f(unsigned short h) {
    return __uint_as_float(((unsigned int)h) << 16);
}
__device__ __forceinline__ v8f wmma_bf16(const v16bf& a, const v16bf& b, const v8f& c) {
    return __builtin_amdgcn_wmma_f32_16x16x32_bf16(false, a, false, b, (short)0, c, false, false);
}
__device__ __forceinline__ float sigmoidf(float x) {
    return 1.0f / (1.0f + __expf(-x));
}

// ---------------- conversion kernels ----------------
__global__ void k_f32_to_bf16(const float* __restrict__ in,
                              unsigned short* __restrict__ out, long n) {
    long i = (long)blockIdx.x * blockDim.x + threadIdx.x;
    long stride = (long)gridDim.x * blockDim.x;
    for (; i < n; i += stride) out[i] = f2bf(in[i]);
}

// W (CONCAT x H) f32 -> Wt (H x CONCAT) bf16, tiled transpose through LDS
__global__ void k_w_transpose_bf16(const float* __restrict__ w,
                                   unsigned short* __restrict__ wt) {
    __shared__ float tile[32][33];
    int k0 = blockIdx.x * 32;   // along CONCAT (38 blocks)
    int n0 = blockIdx.y * 32;   // along H     (32 blocks)
    int tx = threadIdx.x & 31;
    int ty = threadIdx.x >> 5;  // 0..7
#pragma unroll
    for (int r = 0; r < 4; ++r) {
        int row = ty + 8 * r;
        tile[row][tx] = w[(size_t)(k0 + row) * HH + (n0 + tx)];
    }
    __syncthreads();
#pragma unroll
    for (int r = 0; r < 4; ++r) {
        int row = ty + 8 * r;
        wt[(size_t)(n0 + row) * CONCAT + (k0 + tx)] = f2bf(tile[tx][row]);
    }
}

// ---------------- pass 1: z and r gates fused (dual-GEMM) ----------------
// A = [xflat(bf16, MxKF) | hbf(bf16, MxHH)]; B = Wzt/Wrt (HH x CONCAT bf16)
// out: zbf (MxHH bf16), rhbf = sigmoid(r)*h_prev (MxHH bf16)
__global__ __launch_bounds__(256)
void k_gru_pass1(const unsigned short* __restrict__ xflat,
                 const unsigned short* __restrict__ hbf,
                 const unsigned short* __restrict__ wzt,
                 const unsigned short* __restrict__ wrt,
                 const float* __restrict__ bz,
                 const float* __restrict__ br,
                 const float* __restrict__ h_prev,
                 unsigned short* __restrict__ zbf,
                 unsigned short* __restrict__ rhbf) {
    __shared__ __align__(16) unsigned short As[2][128 * 48];     // 2 x (128 M x 32 K)
    __shared__ __align__(16) unsigned short Bs[2][2][64 * 48];   // 2 buf x 2 gate x (64 N x 32 K)

    const int m0 = blockIdx.x * 128;
    const int n0 = blockIdx.y * 64;
    const int t    = threadIdx.x;
    const int wave = t >> 5;
    const int lane = t & 31;
    const int wm = wave & 3;   // 4 wave-rows  -> 128 M
    const int wn = wave >> 2;  // 2 wave-cols  -> 64 N
    const int lm    = lane & 15;
    const int khalf = lane >> 4;

    v8f accz[2][2], accr[2][2];
    const v8f vzero = {0.f, 0.f, 0.f, 0.f, 0.f, 0.f, 0.f, 0.f};
#pragma unroll
    for (int i = 0; i < 2; ++i)
#pragma unroll
        for (int j = 0; j < 2; ++j) { accz[i][j] = vzero; accr[i][j] = vzero; }

    // stage one K-chunk (4 x b128 copies per thread, OOB M rows skipped)
    auto stage = [&](int kc, int buf) {
        const unsigned short* aptr; int ld, koff;
        if (kc < XCHUNKS) { aptr = xflat; ld = KF; koff = kc * KC; }
        else              { aptr = hbf;   ld = HH; koff = (kc - XCHUNKS) * KC; }
#pragma unroll
        for (int i = 0; i < 2; ++i) {
            int idx = t + i * 256;
            int row = idx >> 2, seg = idx & 3;
            if (m0 + row < MM)
                copy_b128(&As[buf][row * 48 + seg * 8],
                          aptr + (size_t)(m0 + row) * ld + koff + seg * 8);
        }
        int nrow = t >> 2, seg = t & 3;
        size_t gsrc = (size_t)(n0 + nrow) * CONCAT + kc * KC + seg * 8;
        copy_b128(&Bs[buf][0][nrow * 48 + seg * 8], wzt + gsrc);
        copy_b128(&Bs[buf][1][nrow * 48 + seg * 8], wrt + gsrc);
    };

    stage(0, 0);
    for (int kc = 0; kc < NKC; ++kc) {
        const int cur = kc & 1;
        __syncthreads();                      // buffer cur^1 fully consumed (iter kc-1)
        if (kc + 1 < NKC) {
            stage(kc + 1, cur ^ 1);           // prefetch next chunk (async, in-flight)
            WAIT_ASYNC(4);                    // chunk kc complete (in-order done)
        } else {
            WAIT_ASYNC(0);
        }
        __syncthreads();                      // chunk kc visible to all waves

        Frag a[2];
#pragma unroll
        for (int ti = 0; ti < 2; ++ti) {
            int r = wm * 32 + ti * 16 + lm;
            a[ti].q[0] = *(const uint4*)(&As[cur][r * 48 + khalf * 8]);
            a[ti].q[1] = *(const uint4*)(&As[cur][r * 48 + 16 + khalf * 8]);
        }
        Frag b[2][2];
#pragma unroll
        for (int g = 0; g < 2; ++g)
#pragma unroll
            for (int tj = 0; tj < 2; ++tj) {
                int n = wn * 32 + tj * 16 + lm;
                b[g][tj].q[0] = *(const uint4*)(&Bs[cur][g][n * 48 + khalf * 8]);
                b[g][tj].q[1] = *(const uint4*)(&Bs[cur][g][n * 48 + 16 + khalf * 8]);
            }
#pragma unroll
        for (int ti = 0; ti < 2; ++ti)
#pragma unroll
            for (int tj = 0; tj < 2; ++tj) {
                accz[ti][tj] = wmma_bf16(a[ti].v, b[0][tj].v, accz[ti][tj]);
                accr[ti][tj] = wmma_bf16(a[ti].v, b[1][tj].v, accr[ti][tj]);
            }
    }

    // epilogue: z = sigmoid(.+bz), rh = sigmoid(.+br)*h_prev
    const int hi = lane >> 4;
#pragma unroll
    for (int ti = 0; ti < 2; ++ti)
#pragma unroll
        for (int tj = 0; tj < 2; ++tj) {
            int colbase = n0 + wn * 32 + tj * 16 + lm;
            float bzc = bz[colbase], brc = br[colbase];
#pragma unroll
            for (int v = 0; v < 8; ++v) {
                int row = m0 + wm * 32 + ti * 16 + v + hi * 8;
                if (row < MM) {
                    size_t o = (size_t)row * HH + colbase;
                    float z = sigmoidf(accz[ti][tj][v] + bzc);
                    float r = sigmoidf(accr[ti][tj][v] + brc);
                    zbf[o]  = f2bf(z);
                    rhbf[o] = f2bf(r * h_prev[o]);
                }
            }
        }
}

// ---------------- pass 2: candidate GEMM + final blend ----------------
__global__ __launch_bounds__(256)
void k_gru_pass2(const unsigned short* __restrict__ xflat,
                 const unsigned short* __restrict__ rhbf,
                 const unsigned short* __restrict__ wct,
                 const float* __restrict__ bc,
                 const unsigned short* __restrict__ zbf,
                 const float* __restrict__ h_prev,
                 float* __restrict__ out) {
    __shared__ __align__(16) unsigned short As[2][128 * 48];
    __shared__ __align__(16) unsigned short Bs[2][64 * 48];

    const int m0 = blockIdx.x * 128;
    const int n0 = blockIdx.y * 64;
    const int t    = threadIdx.x;
    const int wave = t >> 5;
    const int lane = t & 31;
    const int wm = wave & 3;
    const int wn = wave >> 2;
    const int lm    = lane & 15;
    const int khalf = lane >> 4;

    v8f acc[2][2];
    const v8f vzero = {0.f, 0.f, 0.f, 0.f, 0.f, 0.f, 0.f, 0.f};
#pragma unroll
    for (int i = 0; i < 2; ++i)
#pragma unroll
        for (int j = 0; j < 2; ++j) acc[i][j] = vzero;

    auto stage = [&](int kc, int buf) {
        const unsigned short* aptr; int ld, koff;
        if (kc < XCHUNKS) { aptr = xflat; ld = KF; koff = kc * KC; }
        else              { aptr = rhbf;  ld = HH; koff = (kc - XCHUNKS) * KC; }
#pragma unroll
        for (int i = 0; i < 2; ++i) {
            int idx = t + i * 256;
            int row = idx >> 2, seg = idx & 3;
            if (m0 + row < MM)
                copy_b128(&As[buf][row * 48 + seg * 8],
                          aptr + (size_t)(m0 + row) * ld + koff + seg * 8);
        }
        int nrow = t >> 2, seg = t & 3;
        copy_b128(&Bs[buf][nrow * 48 + seg * 8],
                  wct + (size_t)(n0 + nrow) * CONCAT + kc * KC + seg * 8);
    };

    stage(0, 0);
    for (int kc = 0; kc < NKC; ++kc) {
        const int cur = kc & 1;
        __syncthreads();
        if (kc + 1 < NKC) {
            stage(kc + 1, cur ^ 1);
            WAIT_ASYNC(3);
        } else {
            WAIT_ASYNC(0);
        }
        __syncthreads();

        Frag a[2];
#pragma unroll
        for (int ti = 0; ti < 2; ++ti) {
            int r = wm * 32 + ti * 16 + lm;
            a[ti].q[0] = *(const uint4*)(&As[cur][r * 48 + khalf * 8]);
            a[ti].q[1] = *(const uint4*)(&As[cur][r * 48 + 16 + khalf * 8]);
        }
        Frag b[2];
#pragma unroll
        for (int tj = 0; tj < 2; ++tj) {
            int n = wn * 32 + tj * 16 + lm;
            b[tj].q[0] = *(const uint4*)(&Bs[cur][n * 48 + khalf * 8]);
            b[tj].q[1] = *(const uint4*)(&Bs[cur][n * 48 + 16 + khalf * 8]);
        }
#pragma unroll
        for (int ti = 0; ti < 2; ++ti)
#pragma unroll
            for (int tj = 0; tj < 2; ++tj)
                acc[ti][tj] = wmma_bf16(a[ti].v, b[tj].v, acc[ti][tj]);
    }

    const int hi = lane >> 4;
#pragma unroll
    for (int ti = 0; ti < 2; ++ti)
#pragma unroll
        for (int tj = 0; tj < 2; ++tj) {
            int colbase = n0 + wn * 32 + tj * 16 + lm;
            float bcc = bc[colbase];
#pragma unroll
            for (int v = 0; v < 8; ++v) {
                int row = m0 + wm * 32 + ti * 16 + v + hi * 8;
                if (row < MM) {
                    size_t o = (size_t)row * HH + colbase;
                    float hc = tanhf(acc[ti][tj][v] + bcc);
                    float z  = bf2f(zbf[o]);
                    float hp = h_prev[o];
                    out[o] = (1.0f - z) * hp + z * hc;
                }
            }
        }
}

// ---------------- launch ----------------
extern "C" void kernel_launch(void* const* d_in, const int* in_sizes, int n_in,
                              void* d_out, int out_size, void* d_ws, size_t ws_size,
                              hipStream_t stream) {
    const float* diffused_x = (const float*)d_in[0]; // (B,N,K,F) contiguous == (M, 192)
    const float* h_prev     = (const float*)d_in[1]; // (M, 1024)
    const float* Wz = (const float*)d_in[2];
    const float* bz = (const float*)d_in[3];
    const float* Wr = (const float*)d_in[4];
    const float* br = (const float*)d_in[5];
    const float* Wc = (const float*)d_in[6];
    const float* bc = (const float*)d_in[7];
    float* out = (float*)d_out;

    // workspace layout (bf16 buffers), 256-byte aligned regions
    char* ws = (char*)d_ws;
    size_t off = 0;
    auto take = [&](size_t bytes) -> char* {
        char* p = ws + off;
        off = (off + bytes + 255) & ~(size_t)255;
        return p;
    };
    unsigned short* xflat = (unsigned short*)take((size_t)MM * KF * 2);
    unsigned short* hbf   = (unsigned short*)take((size_t)MM * HH * 2);
    unsigned short* rhbf  = (unsigned short*)take((size_t)MM * HH * 2);
    unsigned short* zbf   = (unsigned short*)take((size_t)MM * HH * 2);
    unsigned short* wzt   = (unsigned short*)take((size_t)HH * CONCAT * 2);
    unsigned short* wrt   = (unsigned short*)take((size_t)HH * CONCAT * 2);
    unsigned short* wct   = (unsigned short*)take((size_t)HH * CONCAT * 2);
    (void)ws_size;

    // 1) convert activations to bf16
    {
        long nx = (long)MM * KF;
        k_f32_to_bf16<<<(int)((nx + 1023) / 1024), 1024, 0, stream>>>(diffused_x, xflat, nx);
        long nh = (long)MM * HH;
        k_f32_to_bf16<<<(int)((nh + 1023) / 1024), 1024, 0, stream>>>(h_prev, hbf, nh);
    }
    // 2) convert + transpose weights
    {
        dim3 g(CONCAT / 32, HH / 32);
        k_w_transpose_bf16<<<g, 256, 0, stream>>>(Wz, wzt);
        k_w_transpose_bf16<<<g, 256, 0, stream>>>(Wr, wrt);
        k_w_transpose_bf16<<<g, 256, 0, stream>>>(Wc, wct);
    }
    // 3) z & r gates (fused dual-GEMM)
    {
        dim3 g((MM + 127) / 128, HH / 64);
        k_gru_pass1<<<g, 256, 0, stream>>>(xflat, hbf, wzt, wrt, bz, br, h_prev, zbf, rhbf);
    }
    // 4) candidate GEMM + final blend
    {
        dim3 g((MM + 127) / 128, HH / 64);
        k_gru_pass2<<<g, 256, 0, stream>>>(xflat, rhbf, wct, bc, zbf, h_prev, out);
    }
}
